// EfficientGraphAttentionBlockMoE_76682346103006
// MI455X (gfx1250) — matrix operand order repaired and checked by reading, not compiled, not run on hardware
//
#include <hip/hip_runtime.h>
#include <hip/hip_bf16.h>

// ---------------------------------------------------------------------------
// EfficientGraphAttentionBlockMoE for MI455X (gfx1250, wave32, WMMA bf16)
// Routed top-2 MoE: per-expert GEMMs run only over routed nodes.
// GEMM: 32x128 tile per wave; per 32-deep K-chunk: one clause of 16 b128
// B-fragment loads + 8 b128 A loads, then 16 back-to-back
// v_wmma_f32_16x16x32_bf16.  B pre-packed to bf16 fragment layout.
// ---------------------------------------------------------------------------

#define NN      1024
#define KK      32
#define HH      256
#define DEE     128
#define EE      8
#define NHEADS_ 8
#define NKK     (NN * KK)

typedef __attribute__((ext_vector_type(16))) __bf16 bf16x16;
typedef __attribute__((ext_vector_type(8)))  float  f32x8;

__device__ __forceinline__ float gelu_tanh(float x) {
  const float c = 0.7978845608028654f;  // sqrt(2/pi)
  return 0.5f * x * (1.0f + tanhf(c * (x + 0.044715f * x * x * x)));
}

__device__ __forceinline__ float wave_sum32(float s) {
#pragma unroll
  for (int off = 16; off > 0; off >>= 1) s += __shfl_xor(s, off, 32);
  return s;
}

// Convert two contiguous float4 pairs into one 16-bf16 A fragment.
// A frag (16x32) k order used here, matching pack_b: lane M = lane%16;
//   elems 0..7  <- K = half*16 + 0..7
//   elems 8..15 <- K = half*16 + 8 + 0..7
__device__ __forceinline__ bf16x16 load_a_frag(const float* __restrict__ p, int half) {
  const float4* q0 = (const float4*)(p + (half << 4));
  const float4* q1 = (const float4*)(p + 8 + (half << 4));
  float4 a0 = q0[0], a1 = q0[1], a2 = q1[0], a3 = q1[1];
  bf16x16 f;
  f[0]  = (__bf16)a0.x; f[1]  = (__bf16)a0.y; f[2]  = (__bf16)a0.z; f[3]  = (__bf16)a0.w;
  f[4]  = (__bf16)a1.x; f[5]  = (__bf16)a1.y; f[6]  = (__bf16)a1.z; f[7]  = (__bf16)a1.w;
  f[8]  = (__bf16)a2.x; f[9]  = (__bf16)a2.y; f[10] = (__bf16)a2.z; f[11] = (__bf16)a2.w;
  f[12] = (__bf16)a3.x; f[13] = (__bf16)a3.y; f[14] = (__bf16)a3.z; f[15] = (__bf16)a3.w;
  return f;
}

// One 32-deep K-chunk of the 32x128 tile: load all B frags, then 16 WMMAs.
__device__ __forceinline__ void mma_chunk(f32x8* acc,
                                          const float* __restrict__ Ap0,
                                          const float* __restrict__ Ap1,
                                          int half,
                                          const bf16x16* __restrict__ bbase) {
  bf16x16 aF0 = load_a_frag(Ap0, half);
  bf16x16 aF1 = load_a_frag(Ap1, half);
  bf16x16 bFv[8];
#pragma unroll
  for (int j = 0; j < 8; ++j) bFv[j] = bbase[j * 32];
#pragma unroll
  for (int j = 0; j < 8; ++j) {
    acc[j] = __builtin_amdgcn_wmma_f32_16x16x32_bf16(
        false, aF0, false, bFv[j], (short)0, acc[j], false, false);
    acc[8 + j] = __builtin_amdgcn_wmma_f32_16x16x32_bf16(
        false, aF1, false, bFv[j], (short)0, acc[8 + j], false, false);
  }
}

// ---------------------------------------------------------------------------
// Gating: logits = node_features @ w_gate, top-2 softmax, build routing lists.
// ---------------------------------------------------------------------------
__global__ void gating_kernel(const float* __restrict__ nf,
                              const float* __restrict__ wg,
                              float* __restrict__ gates,
                              int* __restrict__ counts,
                              int* __restrict__ lists) {
  int n = blockIdx.x * blockDim.x + threadIdx.x;
  if (n >= NN) return;
  float lg[EE];
#pragma unroll
  for (int e = 0; e < EE; ++e) lg[e] = 0.f;
  for (int h = 0; h < HH; ++h) {
    float x = nf[(size_t)n * HH + h];
#pragma unroll
    for (int e = 0; e < EE; ++e) lg[e] += x * wg[(size_t)h * EE + e];
  }
  float v1 = -1e30f, v2 = -1e30f;
  int i1 = 0, i2 = 0;
#pragma unroll
  for (int e = 0; e < EE; ++e) {
    float x = lg[e];
    if (x > v1) { v2 = v1; i2 = i1; v1 = x; i1 = e; }
    else if (x > v2) { v2 = x; i2 = e; }
  }
  float e2 = __expf(v2 - v1);
  float s  = 1.f + e2;
#pragma unroll
  for (int e = 0; e < EE; ++e) gates[(size_t)n * EE + e] = 0.f;
  gates[(size_t)n * EE + i1] = 1.f / s;
  gates[(size_t)n * EE + i2] = e2 / s;
  int p1 = atomicAdd(&counts[i1], 1); lists[(size_t)i1 * NN + p1] = n;
  int p2 = atomicAdd(&counts[i2], 1); lists[(size_t)i2 * NN + p2] = n;
}

// ---------------------------------------------------------------------------
// Per-expert row maps for the packed edge-row space r in [0, count*K).
// ---------------------------------------------------------------------------
__global__ void build_maps_kernel(const int* __restrict__ count,
                                  const int* __restrict__ list,
                                  const int* __restrict__ neighbor_list,
                                  int* __restrict__ map_edge,
                                  int* __restrict__ map_send,
                                  int* __restrict__ map_self) {
  int r = blockIdx.x * blockDim.x + threadIdx.x;
  if (r >= (*count) * KK) return;
  int pn = r / KK, k = r % KK;
  int node = list[pn];
  int ge = node * KK + k;
  map_edge[r] = ge;
  map_send[r] = neighbor_list[ge];
  map_self[r] = node;
}

// ---------------------------------------------------------------------------
// Row-wise LayerNorm over H=256 (one wave per row, 8 elems/lane).
// ---------------------------------------------------------------------------
__global__ __launch_bounds__(32) void ln_rows_kernel(const float* __restrict__ x,
                                                     const float* __restrict__ g,
                                                     const float* __restrict__ b,
                                                     float* __restrict__ y) {
  int row = blockIdx.x, lane = threadIdx.x;
  float v[8]; float s = 0.f;
#pragma unroll
  for (int i = 0; i < 8; ++i) { v[i] = x[(size_t)row * HH + lane + i * 32]; s += v[i]; }
  s = wave_sum32(s);
  float mu = s * (1.0f / HH), var = 0.f;
#pragma unroll
  for (int i = 0; i < 8; ++i) { float d = v[i] - mu; var += d * d; }
  var = wave_sum32(var) * (1.0f / HH);
  float inv = rsqrtf(var + 1e-5f);
#pragma unroll
  for (int i = 0; i < 8; ++i) {
    int c = lane + i * 32;
    y[(size_t)row * HH + c] = (v[i] - mu) * inv * g[c] + b[c];
  }
}

// ---------------------------------------------------------------------------
// Pack B [Ktot, Nout] f32 (row-major) into WMMA bf16 B-fragments.
// Fragment f = kt*(Nout/16) + n16 holds 32 lanes x 16 bf16 (32 B/lane):
//   lane elem i <-> K = kt*32 + 16*(lane>>4) + i,  N = n16*16 + (lane&15)
// ---------------------------------------------------------------------------
__global__ void pack_b_kernel(const float* __restrict__ B, int Ktot, int Nout,
                              bf16x16* __restrict__ out) {
  int t = blockIdx.x * blockDim.x + threadIdx.x;
  int nfrag = (Ktot >> 5) * (Nout >> 4);
  if (t >= nfrag * 32) return;
  int frag = t >> 5, lane = t & 31;
  int npern = Nout >> 4;
  int kt = frag / npern, n16 = frag % npern;
  int kbase = (kt << 5) + ((lane >> 4) << 4);
  int col   = (n16 << 4) + (lane & 15);
  bf16x16 v;
#pragma unroll
  for (int i = 0; i < 16; ++i)
    v[i] = (__bf16)B[(size_t)(kbase + i) * Nout + col];
  out[t] = v;
}

// ---------------------------------------------------------------------------
// Generic routed GEMM: C[M, Nout] = act( A[M, Ktot] @ B + bias ).
// A = concat(A0 [first Ka0 cols], A1), each with optional packed-row gather.
// K loop is split per A-source so the hot loop has a fixed base pointer.
// M = (*countPtr) * rowsPerUnit (runtime); tiles past M exit early.
// ---------------------------------------------------------------------------
__global__ __launch_bounds__(32) void gemm_bf16_wmma(
    const float* __restrict__ A0, const int* __restrict__ rows0, int lda0, int Ka0,
    const float* __restrict__ A1, const int* __restrict__ rows1, int lda1,
    int Ktot, const bf16x16* __restrict__ Bp, int Nout,
    const float* __restrict__ bias, float* __restrict__ C,
    const int* __restrict__ countPtr, int rowsPerUnit, int act) {
  const int M = (*countPtr) * rowsPerUnit;
  const int mtile = blockIdx.x;
  if (mtile * 32 >= M) return;
  const int ntile = blockIdx.y;
  const int lane = threadIdx.x;
  const int mloc = lane & 15;
  const int half = lane >> 4;
  const int npern = Nout >> 4;

  // Two M-subtiles per wave; clamp load rows, guard store rows.
  const int mrow0 = mtile * 32 + mloc;
  const int mrow1 = mtile * 32 + 16 + mloc;
  const int ml0 = (mrow0 < M) ? mrow0 : (M - 1);
  const int ml1 = (mrow1 < M) ? mrow1 : (M - 1);

  // Hoist gather lookups out of the K loop.
  const size_t off0_A0 = (size_t)(rows0 ? rows0[ml0] : ml0) * lda0;
  const size_t off1_A0 = (size_t)(rows0 ? rows0[ml1] : ml1) * lda0;
  const size_t off0_A1 = rows1 ? (size_t)rows1[ml0] * lda1 : (size_t)ml0 * lda1;
  const size_t off1_A1 = rows1 ? (size_t)rows1[ml1] * lda1 : (size_t)ml1 * lda1;

  f32x8 z = {0.f, 0.f, 0.f, 0.f, 0.f, 0.f, 0.f, 0.f};
  f32x8 acc[16];
#pragma unroll
  for (int i = 0; i < 16; ++i) acc[i] = z;

  const size_t fragStride = (size_t)npern * 32;        // frags per 32-deep K row
  const bf16x16* bcol = Bp + (size_t)(ntile << 3) * 32 + lane;

  // --- A0 columns ---
  const int kend0 = (Ka0 < Ktot) ? Ka0 : Ktot;
  {
    const float* pa0 = A0 + off0_A0;
    const float* pa1 = A0 + off1_A0;
    for (int kk = 0; kk < kend0; kk += 32) {
      if (kk + 32 < kend0) {
        __builtin_prefetch(pa0 + kk + 32, 0, 1);
        __builtin_prefetch(pa1 + kk + 32, 0, 1);
      }
      mma_chunk(acc, pa0 + kk, pa1 + kk, half, bcol + (size_t)(kk >> 5) * fragStride);
    }
  }
  // --- A1 columns ---
  if (kend0 < Ktot) {
    const float* pa0 = A1 + off0_A1;
    const float* pa1 = A1 + off1_A1;
    for (int kk = kend0; kk < Ktot; kk += 32) {
      int ka = kk - Ka0;
      if (kk + 32 < Ktot) {
        __builtin_prefetch(pa0 + ka + 32, 0, 1);
        __builtin_prefetch(pa1 + ka + 32, 0, 1);
      }
      mma_chunk(acc, pa0 + ka, pa1 + ka, half, bcol + (size_t)(kk >> 5) * fragStride);
    }
  }

  // Epilogue: C layout lane N = lane%16, VGPR v -> M = v + 8*half.
#pragma unroll
  for (int s = 0; s < 2; ++s) {
#pragma unroll
    for (int j = 0; j < 8; ++j) {
      int col = ntile * 128 + j * 16 + mloc;
      float bv = bias ? bias[col] : 0.f;
#pragma unroll
      for (int v = 0; v < 8; ++v) {
        int row = mtile * 32 + s * 16 + (half << 3) + v;
        if (row < M) {
          float x = acc[s * 8 + j][v] + bv;
          if (act) x = gelu_tanh(x);
          C[(size_t)row * Nout + col] = x;
        }
      }
    }
  }
}

// ---------------------------------------------------------------------------
// Windowed multi-head attention: one wave per (packed node, head).
// qkv row layout: [q(256) | k(256) | v(256)]; dh = 32, window K = 32.
// ---------------------------------------------------------------------------
__global__ __launch_bounds__(32) void attn_kernel(const float* __restrict__ qkv,
                                                  const float* __restrict__ attn_mask,
                                                  const int* __restrict__ list,
                                                  const int* __restrict__ count,
                                                  float* __restrict__ attnout) {
  int pn = blockIdx.x;
  if (pn >= *count) return;
  int h = blockIdx.y;
  int lane = threadIdx.x;
  int node = list[pn];

  __shared__ float Ks[KK][33];
  __shared__ float Vs[KK][33];
  const float* base = qkv + (size_t)(pn * KK + lane) * (3 * HH);
#pragma unroll
  for (int d = 0; d < 32; ++d) {
    Ks[lane][d] = base[HH + h * 32 + d];
    Vs[lane][d] = base[2 * HH + h * 32 + d];
  }
  float q[32];
#pragma unroll
  for (int d = 0; d < 32; ++d) q[d] = base[h * 32 + d];  // lane == query row
  __syncthreads();

  const float scale = 0.17677669529663687f;  // 1/sqrt(32)
  float sc[32];
  float mmax = -1e30f;
#pragma unroll
  for (int j = 0; j < 32; ++j) {
    float s = 0.f;
#pragma unroll
    for (int d = 0; d < 32; ++d) s += q[d] * Ks[j][d];
    s = s * scale + attn_mask[(size_t)node * KK + j];
    sc[j] = s;
    mmax = fmaxf(mmax, s);
  }
  float ssum = 0.f;
#pragma unroll
  for (int j = 0; j < 32; ++j) { sc[j] = __expf(sc[j] - mmax); ssum += sc[j]; }
  float inv = 1.0f / ssum;
  float out[32];
#pragma unroll
  for (int d = 0; d < 32; ++d) out[d] = 0.f;
#pragma unroll
  for (int j = 0; j < 32; ++j) {
    float p = sc[j] * inv;
#pragma unroll
    for (int d = 0; d < 32; ++d) out[d] += p * Vs[j][d];
  }
  float* op = attnout + (size_t)(pn * KK + lane) * HH + h * 32;
#pragma unroll
  for (int d = 0; d < 32; ++d) op[d] = out[d];
}

// ---------------------------------------------------------------------------
// ef = edge_out_e + edge_features[map]; efln = LN(ef).  One wave per row.
// ---------------------------------------------------------------------------
__global__ __launch_bounds__(32) void ef_ln_kernel(const float* __restrict__ eo,
                                                   const float* __restrict__ edge_features,
                                                   const int* __restrict__ map_edge,
                                                   const float* __restrict__ g,
                                                   const float* __restrict__ b,
                                                   const int* __restrict__ count,
                                                   float* __restrict__ ef,
                                                   float* __restrict__ efln) {
  int r = blockIdx.x;
  if (r >= (*count) * KK) return;
  int lane = threadIdx.x;
  int ge = map_edge[r];
  float v[8]; float s = 0.f;
#pragma unroll
  for (int i = 0; i < 8; ++i) {
    int c = lane + i * 32;
    float x = eo[(size_t)r * HH + c] + edge_features[(size_t)ge * HH + c];
    ef[(size_t)r * HH + c] = x;
    v[i] = x; s += x;
  }
  s = wave_sum32(s);
  float mu = s * (1.0f / HH), var = 0.f;
#pragma unroll
  for (int i = 0; i < 8; ++i) { float d = v[i] - mu; var += d * d; }
  var = wave_sum32(var) * (1.0f / HH);
  float inv = rsqrtf(var + 1e-5f);
#pragma unroll
  for (int i = 0; i < 8; ++i) {
    int c = lane + i * 32;
    efln[(size_t)r * HH + c] = (v[i] - mu) * inv * g[c] + b[c];
  }
}

// ---------------------------------------------------------------------------
// Masked-mean edge->node aggregate + residual + LN.  One wave per packed node.
// ---------------------------------------------------------------------------
__global__ __launch_bounds__(32) void node_agg_kernel(const float* __restrict__ eo,
                                                      const float* __restrict__ neighbor_mask,
                                                      const float* __restrict__ node_features,
                                                      const int* __restrict__ list,
                                                      const int* __restrict__ count,
                                                      const float* __restrict__ g,
                                                      const float* __restrict__ b,
                                                      float* __restrict__ nfb,
                                                      float* __restrict__ nfln) {
  int pn = blockIdx.x;
  if (pn >= *count) return;
  int lane = threadIdx.x;
  int node = list[pn];
  float acc[8];
#pragma unroll
  for (int i = 0; i < 8; ++i) acc[i] = 0.f;
  float cnt = 1e-5f;
  for (int k = 0; k < KK; ++k) {
    float mk = neighbor_mask[(size_t)node * KK + k];
    cnt += mk;
    const float* row = eo + (size_t)(pn * KK + k) * HH;
#pragma unroll
    for (int i = 0; i < 8; ++i) acc[i] += row[lane + i * 32] * mk;
  }
  float invc = 1.0f / cnt;
  float v[8]; float s = 0.f;
#pragma unroll
  for (int i = 0; i < 8; ++i) {
    int c = lane + i * 32;
    float x = acc[i] * invc + node_features[(size_t)node * HH + c];
    nfb[(size_t)pn * HH + c] = x;
    v[i] = x; s += x;
  }
  s = wave_sum32(s);
  float mu = s * (1.0f / HH), var = 0.f;
#pragma unroll
  for (int i = 0; i < 8; ++i) { float d = v[i] - mu; var += d * d; }
  var = wave_sum32(var) * (1.0f / HH);
  float inv = rsqrtf(var + 1e-5f);
#pragma unroll
  for (int i = 0; i < 8; ++i) {
    int c = lane + i * 32;
    nfln[(size_t)pn * HH + c] = (v[i] - mu) * inv * g[c] + b[c];
  }
}

// ---------------------------------------------------------------------------
// Gate-weighted accumulation into the concatenated output.
// ---------------------------------------------------------------------------
__global__ void edge_accum_kernel(const float* __restrict__ ef,
                                  const float* __restrict__ ffout,
                                  const int* __restrict__ map_edge,
                                  const int* __restrict__ list,
                                  const float* __restrict__ gates,
                                  const int* __restrict__ count,
                                  int e, float* __restrict__ out_edge) {
  int r = blockIdx.x;
  if (r >= (*count) * KK) return;
  int c = threadIdx.x;
  int node = list[r / KK];
  float g = gates[(size_t)node * EE + e];
  out_edge[(size_t)map_edge[r] * HH + c] +=
      g * (ef[(size_t)r * HH + c] + ffout[(size_t)r * HH + c]);
}

__global__ void node_accum_kernel(const float* __restrict__ nfb,
                                  const float* __restrict__ nfout,
                                  const int* __restrict__ list,
                                  const float* __restrict__ gates,
                                  const int* __restrict__ count,
                                  int e, float* __restrict__ out_node) {
  int pn = blockIdx.x;
  if (pn >= *count) return;
  int c = threadIdx.x;
  int node = list[pn];
  float g = gates[(size_t)node * EE + e];
  out_node[(size_t)node * HH + c] +=
      g * (nfb[(size_t)pn * HH + c] + nfout[(size_t)pn * HH + c]);
}

// ---------------------------------------------------------------------------
// Host launcher
// ---------------------------------------------------------------------------
extern "C" void kernel_launch(void* const* d_in, const int* in_sizes, int n_in,
                              void* d_out, int out_size, void* d_ws, size_t ws_size,
                              hipStream_t stream) {
  (void)in_sizes; (void)n_in; (void)ws_size;
  const float* node_features = (const float*)d_in[0];
  const float* edge_features = (const float*)d_in[1];
  const float* edge_raw      = (const float*)d_in[2];
  const int*   neighbor_list = (const int*)  d_in[3];
  const float* neighbor_mask = (const float*)d_in[4];
  const float* attn_mask     = (const float*)d_in[5];
  const float* w_gate        = (const float*)d_in[6];
  const float* W_edge = (const float*)d_in[7];
  const float* b_edge = (const float*)d_in[8];
  const float* W_node = (const float*)d_in[9];
  const float* b_node = (const float*)d_in[10];
  const float* W_msg  = (const float*)d_in[11];
  const float* b_msg  = (const float*)d_in[12];
  const float* W_qkv  = (const float*)d_in[13];
  const float* b_qkv  = (const float*)d_in[14];
  const float* W_out  = (const float*)d_in[15];
  const float* b_out  = (const float*)d_in[16];
  const float* lag = (const float*)d_in[17];
  const float* lab = (const float*)d_in[18];
  const float* lfg = (const float*)d_in[19];
  const float* lfb = (const float*)d_in[20];
  const float* W1n = (const float*)d_in[21];
  const float* b1n = (const float*)d_in[22];
  const float* W2n = (const float*)d_in[23];
  const float* b2n = (const float*)d_in[24];
  const float* W1e = (const float*)d_in[25];
  const float* b1e = (const float*)d_in[26];
  const float* W2e = (const float*)d_in[27];
  const float* b2e = (const float*)d_in[28];

  char* ws = (char*)d_ws;
  size_t o = 0;
  auto take = [&](size_t bytes) -> char* {
    char* p = ws + o;
    o += (bytes + 255) & ~(size_t)255;
    return p;
  };
  float*   gates    = (float*)  take((size_t)NN * EE * 4);
  int*     counts   = (int*)    take((size_t)EE * 4);
  int*     lists    = (int*)    take((size_t)EE * NN * 4);
  int*     map_edge = (int*)    take((size_t)NKK * 4);
  int*     map_send = (int*)    take((size_t)NKK * 4);
  int*     map_self = (int*)    take((size_t)NKK * 4);
  bf16x16* Bpack    = (bf16x16*)take((size_t)512 * 1024);  // >= max Ktot*Nout*2B
  float* nh  = (float*)take((size_t)NN  * HH * 4);
  float* R1  = (float*)take((size_t)NKK * HH * 4);       // eh  -> attnout
  float* R2  = (float*)take((size_t)NKK * HH * 4);       // nodeh -> edge_out_e
  float* R3  = (float*)take((size_t)NKK * HH * 4);       // msg -> ef
  float* R4  = (float*)take((size_t)NKK * 3 * HH * 4);   // qkv -> [efln | ffout | -]
  float* R5  = (float*)take((size_t)NKK * 2 * HH * 4);   // edge FFN hidden
  float* nfb   = (float*)take((size_t)NN * HH * 4);
  float* nfln  = (float*)take((size_t)NN * HH * 4);
  float* nhid  = (float*)take((size_t)NN * 2 * HH * 4);
  float* nfout = (float*)take((size_t)NN * HH * 4);

  float* out_node = (float*)d_out;
  float* out_edge = out_node + (size_t)NN * HH;
  float* efln  = R4;                       // qkv region reuse (dead after attn)
  float* ffout = R4 + (size_t)NKK * HH;

  hipMemsetAsync(d_out, 0, (size_t)out_size * sizeof(float), stream);
  hipMemsetAsync(counts, 0, EE * sizeof(int), stream);
  gating_kernel<<<dim3((NN + 255) / 256), dim3(256), 0, stream>>>(
      node_features, w_gate, gates, counts, lists);

  // pack B then run the routed GEMM (stream order serializes the reuse of Bpack)
  auto gemm = [&](const float* A0, const int* rows0, int lda0, int Ka0,
                  const float* A1, const int* rows1, int lda1,
                  int Ktot, const float* B, int Nout, const float* bias,
                  float* C, const int* cnt, int rowsPerUnit, int act, int mtiles) {
    int nthr = (Ktot >> 5) * (Nout >> 4) * 32;
    pack_b_kernel<<<dim3((nthr + 255) / 256), dim3(256), 0, stream>>>(B, Ktot, Nout, Bpack);
    gemm_bf16_wmma<<<dim3(mtiles, Nout / 128), dim3(32), 0, stream>>>(
        A0, rows0, lda0, Ka0, A1, rows1, lda1, Ktot, Bpack, Nout, bias, C,
        cnt, rowsPerUnit, act);
  };
  const int MT_E = NKK / 32;  // edge-row M tiles (32 rows each)
  const int MT_N = NN / 32;   // node-row M tiles

  for (int e = 0; e < EE; ++e) {
    const int* cnt = counts + e;
    const int* lst = lists + (size_t)e * NN;

    build_maps_kernel<<<dim3(NKK / 256), dim3(256), 0, stream>>>(
        cnt, lst, neighbor_list, map_edge, map_send, map_self);
    ln_rows_kernel<<<dim3(NN), dim3(32), 0, stream>>>(
        node_features, lag + e * HH, lab + e * HH, nh);

    // eh = gelu(edge_raw @ We + be)           [M,128] x [128,256] -> R1
    gemm(edge_raw, map_edge, DEE, DEE, nullptr, nullptr, 0,
         DEE, W_edge + (size_t)e * DEE * HH, HH, b_edge + e * HH,
         R1, cnt, KK, 1, MT_E);
    // nodeh = gelu([nh[send]|nh[self]] @ Wn)  [M,512] x [512,256] -> R2
    gemm(nh, map_send, HH, HH, nh, map_self, HH,
         2 * HH, W_node + (size_t)e * 2 * HH * HH, HH, b_node + e * HH,
         R2, cnt, KK, 1, MT_E);
    // msg = gelu([eh|nodeh] @ Wm)             [M,512] x [512,256] -> R3
    gemm(R1, nullptr, HH, HH, R2, nullptr, HH,
         2 * HH, W_msg + (size_t)e * 2 * HH * HH, HH, b_msg + e * HH,
         R3, cnt, KK, 1, MT_E);
    // qkv = msg @ Wqkv + b                    [M,256] x [256,768] -> R4
    gemm(R3, nullptr, HH, HH, nullptr, nullptr, 0,
         HH, W_qkv + (size_t)e * HH * 3 * HH, 3 * HH, b_qkv + e * 3 * HH,
         R4, cnt, KK, 0, MT_E);
    // attention over the K=32 window          -> R1
    attn_kernel<<<dim3(NN, NHEADS_), dim3(32), 0, stream>>>(
        R4, attn_mask, lst, cnt, R1);
    // edge_out_e = attnout @ Wo + bo          [M,256] x [256,256] -> R2
    gemm(R1, nullptr, HH, HH, nullptr, nullptr, 0,
         HH, W_out + (size_t)e * HH * HH, HH, b_out + e * HH,
         R2, cnt, KK, 0, MT_E);
    // node aggregate + residual + LN          -> nfb, nfln
    node_agg_kernel<<<dim3(NN), dim3(32), 0, stream>>>(
        R2, neighbor_mask, node_features, lst, cnt,
        lfg + e * HH, lfb + e * HH, nfb, nfln);
    // ef = edge_out_e + edge_features; LN     -> R3 (ef), efln
    ef_ln_kernel<<<dim3(NKK), dim3(32), 0, stream>>>(
        R2, edge_features, map_edge, lfg + e * HH, lfb + e * HH, cnt, R3, efln);
    // edge FFN: 256 -> 512 (gelu) -> 256
    gemm(efln, nullptr, HH, HH, nullptr, nullptr, 0,
         HH, W1e + (size_t)e * HH * 2 * HH, 2 * HH, b1e + e * 2 * HH,
         R5, cnt, KK, 1, MT_E);
    gemm(R5, nullptr, 2 * HH, 2 * HH, nullptr, nullptr, 0,
         2 * HH, W2e + (size_t)e * 2 * HH * HH, HH, b2e + e * HH,
         ffout, cnt, KK, 0, MT_E);
    edge_accum_kernel<<<dim3(NKK), dim3(HH), 0, stream>>>(
        R3, ffout, map_edge, lst, gates, cnt, e, out_edge);
    // node FFN: 256 -> 512 (gelu) -> 256
    gemm(nfln, nullptr, HH, HH, nullptr, nullptr, 0,
         HH, W1n + (size_t)e * HH * 2 * HH, 2 * HH, b1n + e * 2 * HH,
         nhid, cnt, 1, 1, MT_N);
    gemm(nhid, nullptr, 2 * HH, 2 * HH, nullptr, nullptr, 0,
         2 * HH, W2n + (size_t)e * 2 * HH * HH, HH, b2n + e * HH,
         nfout, cnt, 1, 0, MT_N);
    node_accum_kernel<<<dim3(NN), dim3(HH), 0, stream>>>(
        nfb, nfout, lst, gates, cnt, e, out_node);
  }
}
